// IsolationEncodingLayer_68307159876091
// MI455X (gfx1250) — matrix operand order repaired and checked by reading, not compiled, not run on hardware
//
#include <hip/hip_runtime.h>
#include <hip/hip_bf16.h>
#include <math.h>

typedef __attribute__((ext_vector_type(16))) _Float16 v16h;
typedef __attribute__((ext_vector_type(4)))  _Float16 v4h;
typedef __attribute__((ext_vector_type(8)))  float    v8f;
typedef __attribute__((ext_vector_type(4)))  unsigned int v4u;

#define KN 131072
#define KM 512
#define KD 128
#define PB 256                 // partial-reduction blocks (rows/block = KN/PB = 512)

// ---------------------------------------------------------------------------
// Row squared-norms only (fallback path for X when ws is small).
// One wave32 per row, float4 loads, xor-shuffle reduction.
// ---------------------------------------------------------------------------
__global__ void __launch_bounds__(256) row_norms_kernel(const float* __restrict__ A,
                                                        float* __restrict__ norms,
                                                        int rows) {
    int wave = (blockIdx.x * blockDim.x + threadIdx.x) >> 5;
    int lane = threadIdx.x & 31;
    if (wave >= rows) return;
    const float4* row = (const float4*)(A + (size_t)wave * KD);
    float4 v = row[lane];                       // KD/4 == 32 == lanes
    float s = v.x * v.x + v.y * v.y + v.z * v.z + v.w * v.w;
#pragma unroll
    for (int off = 16; off >= 1; off >>= 1) s += __shfl_xor(s, off, 32);
    if (lane == 0) norms[wave] = s;
}

// ---------------------------------------------------------------------------
// Fused: row squared-norm + fp32 -> fp16 conversion into workspace.
// One wave32 per row: lane loads float4, stores v4h (8B), shuffle-reduces norm.
// ---------------------------------------------------------------------------
__global__ void __launch_bounds__(256) convert_norm_kernel(const float* __restrict__ A,
                                                           _Float16* __restrict__ Ah,
                                                           float* __restrict__ norms,
                                                           int rows) {
    int wave = (blockIdx.x * blockDim.x + threadIdx.x) >> 5;
    int lane = threadIdx.x & 31;
    if (wave >= rows) return;
    float4 v = ((const float4*)(A + (size_t)wave * KD))[lane];
    v4h h;
    h[0] = (_Float16)v.x; h[1] = (_Float16)v.y;
    h[2] = (_Float16)v.z; h[3] = (_Float16)v.w;
    *(v4h*)(Ah + (size_t)wave * KD + lane * 4) = h;
    float s = v.x * v.x + v.y * v.y + v.z * v.z + v.w * v.w;
#pragma unroll
    for (int off = 16; off >= 1; off >>= 1) s += __shfl_xor(s, off, 32);
    if (lane == 0) norms[wave] = s;
}

// ---------------------------------------------------------------------------
// WMMA distance kernel. Wave -> 16 rows x 64 cols of the dist matrix.
// A fragment (16x32 f16, CDNA5 layout): lane L<16 holds row L, K = k0+{0..7, 16..23};
//   lane L+16 holds row L, K = k0+{8..15, 24..31}.
// B fragment (32x16 f16): lane L<16 holds col L, K = k0+{0..7, 16..23};
//   lane L+16 holds col L, K = k0+{8..15, 24..31}.  Same K order as A.
// XH=true : A loaded directly from pre-converted f16 (zero cvt in hot loop).
// XH=false: A loaded fp32 + converted inline. B always from pre-converted Sh.
// ---------------------------------------------------------------------------
__device__ __forceinline__ void cvt8(v16h& v, int off, float4 a, float4 b) {
    v[off + 0] = (_Float16)a.x; v[off + 1] = (_Float16)a.y;
    v[off + 2] = (_Float16)a.z; v[off + 3] = (_Float16)a.w;
    v[off + 4] = (_Float16)b.x; v[off + 5] = (_Float16)b.y;
    v[off + 6] = (_Float16)b.z; v[off + 7] = (_Float16)b.w;
}

template <bool XH>
__global__ void __launch_bounds__(256) dist_wmma_kernel(const float*    __restrict__ X,
                                                        const _Float16* __restrict__ Xh,
                                                        const _Float16* __restrict__ Sh,
                                                        const float*    __restrict__ x2,
                                                        const float*    __restrict__ s2,
                                                        float*          __restrict__ out) {
    const int lane = threadIdx.x & 31;
    const int gw   = (blockIdx.x * blockDim.x + threadIdx.x) >> 5;   // global wave id
    const int colQuads = KM / 64;                                    // 8
    const int rowTile  = gw / colQuads;            // 8 waves of a block share rowTile
    const int colQuad  = gw % colQuads;
    const int rowBase  = rowTile * 16;
    const int colBase  = colQuad * 64;

    const int sub = lane >> 4;       // half-wave select
    const int lp  = lane & 15;

    v8f acc0 = {}, acc1 = {}, acc2 = {}, acc3 = {};
    const float*    xrow  = X  + (size_t)(rowBase + lp) * KD;
    const _Float16* xrowh = Xh + (size_t)(rowBase + lp) * KD + 8 * sub;

#pragma unroll
    for (int k0 = 0; k0 < KD; k0 += 32) {
        // ---- A fragment ----
        v16h a;
        if (XH) {
            *(v4u*)&a       = *(const v4u*)(xrowh + k0);        // halves 0..7
            *((v4u*)&a + 1) = *(const v4u*)(xrowh + k0 + 16);   // halves 8..15
        } else {
            const int klo = k0 + 8 * sub;
            const float4* p0 = (const float4*)(xrow + klo);
            const float4* p1 = (const float4*)(xrow + klo + 16);
            cvt8(a, 0, p0[0], p0[1]);
            cvt8(a, 8, p1[0], p1[1]);
        }
        // ---- B fragments + WMMA, 4 column tiles ----
#pragma unroll
        for (int t = 0; t < 4; ++t) {
            v16h b;
            const _Float16* sb = Sh + (size_t)(colBase + t * 16 + lp) * KD + k0 + 8 * sub;
            *(v4u*)&b       = *(const v4u*)(sb);
            *((v4u*)&b + 1) = *(const v4u*)(sb + 16);
            v8f c = (t == 0) ? acc0 : (t == 1) ? acc1 : (t == 2) ? acc2 : acc3;
            c = __builtin_amdgcn_wmma_f32_16x16x32_f16(false, a, false, b,
                                                       (short)0, c, false, false);
            if (t == 0) acc0 = c; else if (t == 1) acc1 = c;
            else if (t == 2) acc2 = c; else acc3 = c;
        }
    }

    // ---- epilogue: d = sqrt(max(x2 + s2 - 2*acc, 0)) ----
    float s2v0 = s2[colBase +  0 + lp];
    float s2v1 = s2[colBase + 16 + lp];
    float s2v2 = s2[colBase + 32 + lp];
    float s2v3 = s2[colBase + 48 + lp];

#pragma unroll
    for (int r = 0; r < 8; ++r) {
        const int row = rowBase + r + 8 * sub;      // C/D layout: half-wave selects M / M+8
        const float xv = x2[row];
        float* o = out + (size_t)row * KM + colBase + lp;
        o[ 0] = sqrtf(fmaxf(xv + s2v0 - 2.0f * acc0[r], 0.0f));
        o[16] = sqrtf(fmaxf(xv + s2v1 - 2.0f * acc1[r], 0.0f));
        o[32] = sqrtf(fmaxf(xv + s2v2 - 2.0f * acc2[r], 0.0f));
        o[48] = sqrtf(fmaxf(xv + s2v3 - 2.0f * acc3[r], 0.0f));
    }
}

// ---------------------------------------------------------------------------
// Online per-column (min, sum exp(min - v)) partials. thread = column.
// ---------------------------------------------------------------------------
__global__ void __launch_bounds__(512) col_partials_kernel(const float* __restrict__ dist,
                                                           float* __restrict__ partials,
                                                           int rowsPerBlock) {
    const int col = threadIdx.x;                  // blockDim.x == KM
    const size_t r0 = (size_t)blockIdx.x * rowsPerBlock;

    float mn = dist[r0 * KM + col];
    float ss = 1.0f;
    for (int r = 1; r < rowsPerBlock; ++r) {
        float v = dist[(r0 + r) * KM + col];
        if (v < mn) {
            ss = ss * __expf(v - mn) + 1.0f;
            mn = v;
        } else {
            ss += __expf(mn - v);
        }
    }
    float* p = partials + (size_t)blockIdx.x * 2 * KM;
    p[col]      = mn;
    p[KM + col] = ss;
}

// ---------------------------------------------------------------------------
// Combine PB partials per column -> colstat: [0,KM)=min, [KM,2KM)=1/sum.
// ---------------------------------------------------------------------------
__global__ void __launch_bounds__(256) col_combine_kernel(const float* __restrict__ partials,
                                                          float* __restrict__ colstat) {
    int col = blockIdx.x * blockDim.x + threadIdx.x;
    if (col >= KM) return;
    float mn = partials[col];
    float ss = partials[KM + col];
    for (int b = 1; b < PB; ++b) {
        const float* p = partials + (size_t)b * 2 * KM;
        float m2 = p[col];
        float s2 = p[KM + col];
        float mnew = fminf(mn, m2);
        ss = ss * __expf(mnew - mn) + s2 * __expf(mnew - m2);
        mn = mnew;
    }
    colstat[col]      = mn;
    colstat[KM + col] = 1.0f / ss;
}

// ---------------------------------------------------------------------------
// out[i] = exp(min[col] - dist[i]) * invsum[col], in place.
// ---------------------------------------------------------------------------
__global__ void __launch_bounds__(256) finalize_kernel(float* __restrict__ out,
                                                       const float* __restrict__ colstat) {
    size_t i = (size_t)blockIdx.x * blockDim.x + threadIdx.x;
    int col = (int)(i & (KM - 1));
    float v = out[i];
    out[i] = __expf(colstat[col] - v) * colstat[KM + col];
}

// ---------------------------------------------------------------------------
extern "C" void kernel_launch(void* const* d_in, const int* in_sizes, int n_in,
                              void* d_out, int out_size, void* d_ws, size_t ws_size,
                              hipStream_t stream) {
    const float* X = (const float*)d_in[0];   // [KN, KD]
    const float* S = (const float*)d_in[1];   // [KM, KD]
    float* out = (float*)d_out;               // [KN, KM]

    float* ws       = (float*)d_ws;
    float* x2       = ws;                                   // KN floats
    float* s2       = x2 + KN;                              // KM floats
    float* partials = s2 + KM;                              // PB*2*KM floats
    float* colstat  = partials + (size_t)PB * 2 * KM;       // 2*KM floats
    size_t base_floats = (size_t)KN + KM + (size_t)PB * 2 * KM + 2 * KM;   // ~395k floats

    _Float16* Sh = (_Float16*)(ws + base_floats);           // KM*KD halves (128 KB)
    size_t sh_floats = (size_t)KM * KD / 2;
    _Float16* Xh = (_Float16*)(ws + base_floats + sh_floats);  // KN*KD halves (32 MB)

    size_t need_base = (base_floats + sh_floats) * sizeof(float);
    size_t need_xh   = need_base + (size_t)KN * KD * sizeof(_Float16);
    const bool useXh = (ws_size >= need_xh);   // deterministic: ws_size fixed per run

    // 1) norms (+ f16 pre-conversion)
    convert_norm_kernel<<<KM / 8, 256, 0, stream>>>(S, Sh, s2, KM);
    if (useXh) {
        convert_norm_kernel<<<KN / 8, 256, 0, stream>>>(X, Xh, x2, KN);
    } else {
        row_norms_kernel<<<KN / 8, 256, 0, stream>>>(X, x2, KN);
    }

    // 2) WMMA distance matrix: waves = (KN/16)*(KM/64) = 65536 -> 8192 blocks of 8 waves
    const int gemmBlocks = (KN / 16) * (KM / 64) / 8;
    if (useXh) {
        dist_wmma_kernel<true ><<<gemmBlocks, 256, 0, stream>>>(X, Xh, Sh, x2, s2, out);
    } else {
        dist_wmma_kernel<false><<<gemmBlocks, 256, 0, stream>>>(X, Xh, Sh, x2, s2, out);
    }

    // 3) column-wise online softmax partials
    col_partials_kernel<<<PB, KM, 0, stream>>>(out, partials, KN / PB);

    // 4) combine partials
    col_combine_kernel<<<(KM + 255) / 256, 256, 0, stream>>>(partials, colstat);

    // 5) normalize in place
    finalize_kernel<<<(int)(((size_t)KN * KM) / 256), 256, 0, stream>>>(out, colstat);
}
